// ContextTransformerLayer_76630806495867
// MI455X (gfx1250) — compile-verified, matrix operand order checked by reading, not run on hardware
//
#include <hip/hip_runtime.h>
#include <hip/hip_bf16.h>
#include <math.h>

// ---------------------------------------------------------------------------
// Transformer encoder layer (pre-norm, MHA + GELU FFN) for MI455X / gfx1250.
// All matmuls run on v_wmma_f32_16x16x32_bf16 (wave32 WMMA).
// GEMM A- and B-tiles are staged global->LDS by the Tensor Data Mover
// (tensor_load_to_lds + s_wait_tensorcnt) with double-buffered LDS;
// weights are pre-transposed (and converted to bf16) so both tiles are
// contiguous-in-K 2D TDM descriptors.
// ---------------------------------------------------------------------------

typedef __bf16 bf16;
typedef bf16  v8bf  __attribute__((ext_vector_type(8)));
typedef bf16  v16bf __attribute__((ext_vector_type(16)));
typedef float v8f   __attribute__((ext_vector_type(8)));
typedef unsigned int u32x4 __attribute__((ext_vector_type(4)));
typedef int   i32x4 __attribute__((ext_vector_type(4)));
typedef int   i32x8 __attribute__((ext_vector_type(8)));

#define WMMA_BF16(a, b, c) \
  __builtin_amdgcn_wmma_f32_16x16x32_bf16(false, (a), false, (b), (short)0, (c), false, false)

// Load a 16-lane bf16 A/B fragment for 16x16x32 WMMA.
// Per CDNA5 ISA layout (16-bit A-matrix 16x32): lane r = lane&15 holds row r,
// with K = {koff..koff+7, koff+16..koff+23}, koff = (lane>>4)*8.
__device__ __forceinline__ v16bf frag16(const bf16* base, int koff) {
  v8bf lo = *(const v8bf*)(base + koff);
  v8bf hi = *(const v8bf*)(base + koff + 16);
  return __builtin_shufflevector(lo, hi, 0, 1, 2, 3, 4, 5, 6, 7,
                                 8, 9, 10, 11, 12, 13, 14, 15);
}

// LDS byte offset of a __shared__ object: on AMDGCN the generic pointer to
// LDS is {shared-aperture, offset}, so the low 32 bits are the LDS offset.
__device__ __forceinline__ unsigned lds_offset(const void* p) {
  return (unsigned)(unsigned long long)(uintptr_t)p;
}

// Issue a TDM 2D tile load: rows x cols elements of bf16, row-major source
// with row stride `stride` (elements), packed into LDS at byte offset
// `lds_addr`. D# bit packing per CDNA5 ISA 8.3/8.4.
__device__ __forceinline__ void tdm_load_tile_bf16(
    const bf16* gptr, unsigned lds_addr, int cols /*tile_dim0*/,
    int rows /*tile_dim1*/, int stride /*elements*/, int tensor_rows) {
  unsigned long long ga = (unsigned long long)(uintptr_t)gptr;
  u32x4 g0;
  g0[0] = 1u;                                   // count=1, user descriptor
  g0[1] = lds_addr;                             // lds_addr (bytes)
  g0[2] = (unsigned)ga;                         // global_addr[31:0]
  g0[3] = (unsigned)((ga >> 32) & 0x01FFFFFFu)  // global_addr[56:32]
          | (2u << 30);                         // type = 2 ("image")
  i32x8 g1;
  g1[0] = (int)(1u << 16);                      // wg_mask=0, data_size=1 (2B)
  g1[1] = (int)(((unsigned)stride & 0xFFFFu) << 16);        // tensor_dim0 lo16
  g1[2] = (int)((((unsigned)stride >> 16) & 0xFFFFu)        // tensor_dim0 hi16
          | (((unsigned)tensor_rows & 0xFFFFu) << 16));     // tensor_dim1 lo16
  g1[3] = (int)((((unsigned)tensor_rows >> 16) & 0xFFFFu)   // tensor_dim1 hi16
          | (((unsigned)cols & 0xFFFFu) << 16));            // tile_dim0
  g1[4] = (int)((unsigned)rows & 0xFFFFu);                  // tile_dim1 (tile_dim2=0)
  g1[5] = (int)(unsigned)stride;                            // tensor_dim0_stride lo32
  g1[6] = 0;                                                // stride hi, dim1_stride lo
  g1[7] = 0;
  i32x4 gz = {0, 0, 0, 0};
#if defined(__clang_major__) && (__clang_major__ >= 23)
  i32x8 gz8 = {0, 0, 0, 0, 0, 0, 0, 0};
  __builtin_amdgcn_tensor_load_to_lds(g0, g1, gz, gz, gz8, 0);
#else
  __builtin_amdgcn_tensor_load_to_lds(g0, g1, gz, gz, 0);
#endif
}

// ---------------------------------------------------------------------------
// f32 [K,N] -> bf16 transposed [N,K] (weights), LDS-tiled 32x32.
// ---------------------------------------------------------------------------
__global__ __launch_bounds__(256) void cvt_transpose_kernel(
    const float* __restrict__ in, bf16* __restrict__ out, int K, int N) {
  __shared__ float tile[32][33];
  const int k0 = blockIdx.y * 32;
  const int n0 = blockIdx.x * 32;
  const int tx = threadIdx.x & 31;
  const int ty = threadIdx.x >> 5;  // 0..7
#pragma unroll
  for (int i = 0; i < 32; i += 8)
    tile[ty + i][tx] = in[(size_t)(k0 + ty + i) * N + n0 + tx];
  __syncthreads();
#pragma unroll
  for (int i = 0; i < 32; i += 8)
    out[(size_t)(n0 + ty + i) * K + k0 + tx] = (bf16)tile[tx][ty + i];
}

// ---------------------------------------------------------------------------
// LayerNorm over D=1024: one 256-thread block per row, bf16 output.
// ---------------------------------------------------------------------------
__global__ __launch_bounds__(256) void layernorm_kernel(
    const float* __restrict__ x, const float* __restrict__ g,
    const float* __restrict__ b, bf16* __restrict__ out) {
  const int row = blockIdx.x;
  const int t = threadIdx.x;
  const float* xr = x + (size_t)row * 1024;
  __shared__ float red[512];
  float vals[4];
  float s = 0.f, s2 = 0.f;
#pragma unroll
  for (int i = 0; i < 4; i++) {
    float v = xr[t + i * 256];
    vals[i] = v;
    s += v;
    s2 += v * v;
  }
  red[t] = s;
  red[256 + t] = s2;
  __syncthreads();
  for (int off = 128; off > 0; off >>= 1) {
    if (t < off) {
      red[t] += red[t + off];
      red[256 + t] += red[256 + t + off];
    }
    __syncthreads();
  }
  const float mu = red[0] * (1.0f / 1024.0f);
  const float var = red[256] * (1.0f / 1024.0f) - mu * mu;
  const float inv = rsqrtf(var + 1e-5f);
  bf16* orow = out + (size_t)row * 1024;
#pragma unroll
  for (int i = 0; i < 4; i++) {
    int c = t + i * 256;
    orow[c] = (bf16)((vals[i] - mu) * inv * g[c] + b[c]);
  }
}

// ---------------------------------------------------------------------------
// bf16 WMMA GEMM, fully TDM-fed: C[M,N] = A[M,K] * Wt[N,K]^T + bias.
// Epilogue modes:
//   mode 0: outB = bf16(C)
//   mode 1: outF = resid + C                (f32)
//   mode 2: outB = bf16(gelu_exact(C))
// Block tile 128x128, K-step 32, double-buffered LDS. 256 threads = 8 waves;
// wave 0 issues two tensor_load_to_lds per K-step (next iteration's tiles)
// and synchronizes with s_wait_tensorcnt; all waves run 4x2 WMMA tiles.
// ---------------------------------------------------------------------------
__global__ __launch_bounds__(256) void gemm_bf16_wmma(
    const bf16* __restrict__ A, const bf16* __restrict__ Wt,
    const float* __restrict__ bias, const float* __restrict__ resid,
    bf16* __restrict__ outB, float* __restrict__ outF,
    int M, int N, int K, int mode) {
  __shared__ bf16 As[2][128][32];   // [buf][m][k]
  __shared__ bf16 Bs[2][128][32];   // [buf][n][k]

  const int t = threadIdx.x;
  const int lane = t & 31;
  const int w = t >> 5;
  const int wm = (w >> 2) * 64;  // wave row offset inside block tile
  const int wn = (w & 3) * 32;   // wave col offset inside block tile
  const int m0 = blockIdx.y * 128;
  const int n0 = blockIdx.x * 128;
  const int lr = (lane >> 4) << 3;  // 0 or 8 (k-offset / M-offset selector)
  const int lc = lane & 15;
  const unsigned lds_as[2] = {lds_offset(&As[0][0][0]), lds_offset(&As[1][0][0])};
  const unsigned lds_bs[2] = {lds_offset(&Bs[0][0][0]), lds_offset(&Bs[1][0][0])};
  const bf16* arow = &A[(size_t)m0 * K];
  const bf16* brow = &Wt[(size_t)n0 * K];

  const v8f zero8 = {0.f, 0.f, 0.f, 0.f, 0.f, 0.f, 0.f, 0.f};
  v8f acc[4][2];
#pragma unroll
  for (int mi = 0; mi < 4; mi++)
#pragma unroll
    for (int ni = 0; ni < 2; ni++) acc[mi][ni] = zero8;

  // prologue: stage tiles for iteration 0
  if (w == 0) {
    tdm_load_tile_bf16(arow, lds_as[0], 32, 128, K, M - m0);
    tdm_load_tile_bf16(brow, lds_bs[0], 32, 128, K, N - n0);
  }

  const int nk = K >> 5;
  for (int i = 0; i < nk; i++) {
    const int cur = i & 1;
    if (w == 0) {
      if (i + 1 < nk) {
        // prefetch next K-step into the other buffer, then wait for the
        // current pair (tensor ops complete in order per wave).
        const int k1 = (i + 1) << 5;
        tdm_load_tile_bf16(arow + k1, lds_as[cur ^ 1], 32, 128, K, M - m0);
        tdm_load_tile_bf16(brow + k1, lds_bs[cur ^ 1], 32, 128, K, N - n0);
        __builtin_amdgcn_s_wait_tensorcnt(2);
      } else {
        __builtin_amdgcn_s_wait_tensorcnt(0);
      }
    }
    __syncthreads();

    v16bf af[4], bfr[2];
#pragma unroll
    for (int mi = 0; mi < 4; mi++) af[mi] = frag16(&As[cur][wm + mi * 16 + lc][0], lr);
#pragma unroll
    for (int ni = 0; ni < 2; ni++) bfr[ni] = frag16(&Bs[cur][wn + ni * 16 + lc][0], lr);

#pragma unroll
    for (int mi = 0; mi < 4; mi++)
#pragma unroll
      for (int ni = 0; ni < 2; ni++)
        acc[mi][ni] = WMMA_BF16(af[mi], bfr[ni], acc[mi][ni]);

    __syncthreads();
  }

  // ---- epilogue. Acc layout: VGPR j -> M = j + lr (lanes>=16 add 8), N = lc.
#pragma unroll
  for (int mi = 0; mi < 4; mi++) {
#pragma unroll
    for (int ni = 0; ni < 2; ni++) {
      const int col = n0 + wn + ni * 16 + lc;
      const float bv = bias[col];
#pragma unroll
      for (int j = 0; j < 8; j++) {
        const int row = m0 + wm + mi * 16 + lr + j;
        const size_t o = (size_t)row * N + col;
        float v = acc[mi][ni][j] + bv;
        if (mode == 1) {
          outF[o] = resid[o] + v;
        } else if (mode == 2) {
          float ge = 0.5f * v * (1.0f + erff(v * 0.70710678118654752f));
          outB[o] = (bf16)ge;
        } else {
          outB[o] = (bf16)v;
        }
      }
    }
  }
}

// ---------------------------------------------------------------------------
// Flash attention: one block per (128 queries, head, batch); 8 waves, each
// wave owns a 16-query tile. Streams K/V in 32-key tiles through LDS.
// K tile staged by TDM; V tile staged transposed by all threads.
// Q/K/V/O layout: [B*S, D] bf16 with head h at columns h*64 .. h*64+63.
// ---------------------------------------------------------------------------
__global__ __launch_bounds__(256) void attn_wmma(
    const bf16* __restrict__ Q, const bf16* __restrict__ Kg,
    const bf16* __restrict__ V, bf16* __restrict__ O, int S, int D) {
  __shared__ bf16 Ks[32][64];       // [key][hd]  (TDM destination)
  __shared__ bf16 Vt[64][32];       // [hd][key]  (transposed for B fragments)
  __shared__ bf16 Pb[8][16][32];    // per-wave P tile staging

  const int t = threadIdx.x;
  const int lane = t & 31;
  const int w = t >> 5;
  const int b = blockIdx.z;
  const int h = blockIdx.y;
  const int qbase = blockIdx.x * 128 + w * 16;
  const int lr = (lane >> 4) << 3;
  const int lc = lane & 15;
  const unsigned lds_ks = lds_offset(&Ks[0][0]);

  // Q fragments for this wave's 16 queries (K-dim = 64 -> two 16x32 frags)
  const size_t qrow = (size_t)(b * S + qbase + lc) * D + h * 64;
  const v16bf aq0 = frag16(&Q[qrow], lr);       // hd 0..31
  const v16bf aq1 = frag16(&Q[qrow + 32], lr);  // hd 32..63

  const v8f zero8 = {0.f, 0.f, 0.f, 0.f, 0.f, 0.f, 0.f, 0.f};
  v8f o0 = zero8, o1 = zero8, o2 = zero8, o3 = zero8;
  float m[8], l[8];
#pragma unroll
  for (int j = 0; j < 8; j++) { m[j] = -1e30f; l[j] = 0.f; }
  const float sc = 0.125f;  // 1/sqrt(64)

  for (int kb = 0; kb < S; kb += 32) {
    // ---- K tile (32 keys x 64 hd) via TDM; V tile transposed manually
    if (w == 0) {
      tdm_load_tile_bf16(&Kg[(size_t)(b * S + kb) * D + h * 64], lds_ks,
                         /*cols=*/64, /*rows=*/32, /*stride=*/D,
                         /*tensor_rows=*/S - kb);
    }
    {
      int r = t >> 3, c = (t & 7) << 3;
      v8bf tv = *(const v8bf*)&V[(size_t)(b * S + kb + r) * D + h * 64 + c];
#pragma unroll
      for (int e = 0; e < 8; e++) Vt[c + e][r] = tv[e];
    }
    if (w == 0) __builtin_amdgcn_s_wait_tensorcnt(0);
    __syncthreads();

    // ---- scores S = Q * K^T  (two 16x16 key tiles, K-dim 64)
    v8f s0 = zero8, s1 = zero8;
    s0 = WMMA_BF16(aq0, frag16(&Ks[lc][0], lr), s0);
    s0 = WMMA_BF16(aq1, frag16(&Ks[lc][32], lr), s0);
    s1 = WMMA_BF16(aq0, frag16(&Ks[16 + lc][0], lr), s1);
    s1 = WMMA_BF16(aq1, frag16(&Ks[16 + lc][32], lr), s1);

    // ---- streaming softmax update (row stats via width-16 xor shuffles)
#pragma unroll
    for (int j = 0; j < 8; j++) {
      float p0 = s0[j] * sc, p1 = s1[j] * sc;
      float mx = fmaxf(p0, p1);
      for (int off = 1; off < 16; off <<= 1) mx = fmaxf(mx, __shfl_xor(mx, off, 16));
      float mn = fmaxf(m[j], mx);
      float r = __expf(m[j] - mn);
      m[j] = mn;
      float e0 = __expf(p0 - mn), e1 = __expf(p1 - mn);
      float rs = e0 + e1;
      for (int off = 1; off < 16; off <<= 1) rs += __shfl_xor(rs, off, 16);
      l[j] = l[j] * r + rs;
      o0[j] *= r; o1[j] *= r; o2[j] *= r; o3[j] *= r;
      Pb[w][lr + j][lc] = (bf16)e0;
      Pb[w][lr + j][16 + lc] = (bf16)e1;
    }
    // wave-private LDS RAW: make the P stores visible before fragment reload
    asm volatile("s_wait_dscnt 0x0" ::: "memory");

    // ---- ctx += P(16x32) * V(32x64): one A frag, four 16-col B frags
    v16bf ap = frag16(&Pb[w][lc][0], lr);
    o0 = WMMA_BF16(ap, frag16(&Vt[lc][0], lr), o0);
    o1 = WMMA_BF16(ap, frag16(&Vt[16 + lc][0], lr), o1);
    o2 = WMMA_BF16(ap, frag16(&Vt[32 + lc][0], lr), o2);
    o3 = WMMA_BF16(ap, frag16(&Vt[48 + lc][0], lr), o3);

    __syncthreads();
  }

  // ---- normalize and store ctx (bf16)
#pragma unroll
  for (int j = 0; j < 8; j++) {
    const float inv = 1.0f / l[j];
    const size_t rbase = (size_t)(b * S + qbase + lr + j) * D + h * 64;
    O[rbase + lc] = (bf16)(o0[j] * inv);
    O[rbase + 16 + lc] = (bf16)(o1[j] * inv);
    O[rbase + 32 + lc] = (bf16)(o2[j] * inv);
    O[rbase + 48 + lc] = (bf16)(o3[j] * inv);
  }
}

// ---------------------------------------------------------------------------
// Host orchestration
// ---------------------------------------------------------------------------
extern "C" void kernel_launch(void* const* d_in, const int* in_sizes, int n_in,
                              void* d_out, int out_size, void* d_ws, size_t ws_size,
                              hipStream_t stream) {
  (void)in_sizes; (void)n_in; (void)out_size; (void)ws_size;
  const int B = 8, S = 1024, D = 1024, H = 16, FF = 4096;
  const int M = B * S;

  const float* x  = (const float*)d_in[0];
  const float* Wq = (const float*)d_in[1];  const float* bq = (const float*)d_in[2];
  const float* Wk = (const float*)d_in[3];  const float* bk = (const float*)d_in[4];
  const float* Wv = (const float*)d_in[5];  const float* bv = (const float*)d_in[6];
  const float* Wo = (const float*)d_in[7];  const float* bo = (const float*)d_in[8];
  const float* W1 = (const float*)d_in[9];  const float* b1 = (const float*)d_in[10];
  const float* W2 = (const float*)d_in[11]; const float* b2 = (const float*)d_in[12];
  const float* g1 = (const float*)d_in[13]; const float* be1 = (const float*)d_in[14];
  const float* g2 = (const float*)d_in[15]; const float* be2 = (const float*)d_in[16];

  char* ws = (char*)d_ws;
  size_t off = 0;
  auto alloc = [&](size_t bytes) -> void* {
    void* p = ws + off;
    off += (bytes + 255) & ~(size_t)255;
    return p;
  };
  // transposed bf16 weights: wq_t[N][K] etc.
  bf16* wq_t = (bf16*)alloc((size_t)D * D * 2);
  bf16* wk_t = (bf16*)alloc((size_t)D * D * 2);
  bf16* wv_t = (bf16*)alloc((size_t)D * D * 2);
  bf16* wo_t = (bf16*)alloc((size_t)D * D * 2);
  bf16* w1_t = (bf16*)alloc((size_t)D * FF * 2);   // [FF][D]
  bf16* w2_t = (bf16*)alloc((size_t)FF * D * 2);   // [D][FF]
  bf16* hbf  = (bf16*)alloc((size_t)M * D * 2);    // LN output (reused for LN2)
  bf16* qbf  = (bf16*)alloc((size_t)M * D * 2);
  bf16* kbf  = (bf16*)alloc((size_t)M * D * 2);
  bf16* vbf  = (bf16*)alloc((size_t)M * D * 2);
  bf16* ctxb = (bf16*)alloc((size_t)M * D * 2);
  float* x1  = (float*)alloc((size_t)M * D * 4);
  bf16* act  = qbf;  // reuse q/k/v/ctx (4 x 16MB contiguous = M*FF bf16)

  // 1) weights -> bf16, transposed to [N][K]
  dim3 tb(256);
  cvt_transpose_kernel<<<dim3(D / 32, D / 32), tb, 0, stream>>>(Wq, wq_t, D, D);
  cvt_transpose_kernel<<<dim3(D / 32, D / 32), tb, 0, stream>>>(Wk, wk_t, D, D);
  cvt_transpose_kernel<<<dim3(D / 32, D / 32), tb, 0, stream>>>(Wv, wv_t, D, D);
  cvt_transpose_kernel<<<dim3(D / 32, D / 32), tb, 0, stream>>>(Wo, wo_t, D, D);
  cvt_transpose_kernel<<<dim3(FF / 32, D / 32), tb, 0, stream>>>(W1, w1_t, D, FF);
  cvt_transpose_kernel<<<dim3(D / 32, FF / 32), tb, 0, stream>>>(W2, w2_t, FF, D);

  // 2) LN1
  layernorm_kernel<<<M, 256, 0, stream>>>(x, g1, be1, hbf);

  // 3) QKV projections
  dim3 gd(D / 128, M / 128);
  gemm_bf16_wmma<<<gd, 256, 0, stream>>>(hbf, wq_t, bq, nullptr, qbf, nullptr, M, D, D, 0);
  gemm_bf16_wmma<<<gd, 256, 0, stream>>>(hbf, wk_t, bk, nullptr, kbf, nullptr, M, D, D, 0);
  gemm_bf16_wmma<<<gd, 256, 0, stream>>>(hbf, wv_t, bv, nullptr, vbf, nullptr, M, D, D, 0);

  // 4) attention
  dim3 ga(S / 128, H, B);
  attn_wmma<<<ga, 256, 0, stream>>>(qbf, kbf, vbf, ctxb, S, D);

  // 5) output projection + residual -> x1 (f32)
  gemm_bf16_wmma<<<gd, 256, 0, stream>>>(ctxb, wo_t, bo, x, nullptr, x1, M, D, D, 1);

  // 6) LN2
  layernorm_kernel<<<M, 256, 0, stream>>>(x1, g2, be2, hbf);

  // 7) FFN up + exact GELU
  dim3 gf1(FF / 128, M / 128);
  gemm_bf16_wmma<<<gf1, 256, 0, stream>>>(hbf, w1_t, b1, nullptr, act, nullptr, M, FF, D, 2);

  // 8) FFN down + residual -> d_out (f32)
  dim3 gf2(D / 128, M / 128);
  gemm_bf16_wmma<<<gf2, 256, 0, stream>>>(act, w2_t, b2, x1, nullptr, (float*)d_out, M, D, FF, 1);
}
